// SimpleSOLOV2Decoder_63891933495892
// MI455X (gfx1250) — 8-bit matrix operand order repaired and checked by reading, compiled, not run on hardware
//
#include <hip/hip_runtime.h>
#include <hip/hip_bf16.h>
#include <float.h>

#define NINST   500
#define NPAD    512
#define HH      200
#define WW      304
#define HW      60800
#define KDIM    HW
#define SIGMA_F 2.0f
#define THR_F   0.5f

typedef int v8i __attribute__((ext_vector_type(8)));
typedef int v4i __attribute__((ext_vector_type(4)));
typedef int v2i __attribute__((ext_vector_type(2)));

__global__ void k_threshold(const float* __restrict__ preds,
                            unsigned char* __restrict__ Mq,
                            int* __restrict__ areaI) {
    const int row = blockIdx.x;
    const int tid = threadIdx.x;
    __shared__ int red[256];
    int cnt = 0;
    unsigned char* dst = Mq + (size_t)row * KDIM;
    if (row < NINST) {
        const float* src = preds + (size_t)row * HW;
        for (int hw = tid; hw < HW; hw += 256) {
            unsigned char v = (src[hw] > THR_F) ? 1u : 0u;
            dst[hw] = v;
            cnt += v;
        }
    } else {
        for (int hw = tid; hw < HW; hw += 256) dst[hw] = 0;
    }
    red[tid] = cnt;
    __syncthreads();
    for (int s = 128; s > 0; s >>= 1) {
        if (tid < s) red[tid] += red[tid + s];
        __syncthreads();
    }
    if (tid == 0) areaI[row] = red[0];
}

__global__ void k_sort_scores(const float* __restrict__ scores,
                              int* __restrict__ order) {
    const int tid = threadIdx.x;
    __shared__ float ss[NPAD];
    ss[tid] = (tid < NINST) ? scores[tid] : -FLT_MAX;
    __syncthreads();
    if (tid < NINST) {
        const float sj = ss[tid];
        int rank = 0;
        for (int k = 0; k < NINST; ++k) {
            float sk = ss[k];
            rank += (sk > sj) || (sk == sj && k < tid);
        }
        order[rank] = tid;
    }
}

__device__ __forceinline__ v8i load_fragA(const unsigned char* rowPtr,
                                          int k0, int half) {
    const unsigned char* ap = rowPtr + k0 + half * 16;
    v2i p0 = *reinterpret_cast<const v2i*>(ap);
    v2i p1 = *reinterpret_cast<const v2i*>(ap + 8);
    v2i p2 = *reinterpret_cast<const v2i*>(ap + 32);
    v2i p3 = *reinterpret_cast<const v2i*>(ap + 40);
    return (v8i){p0[0], p0[1], p1[0], p1[1], p2[0], p2[1], p3[0], p3[1]};
}

__device__ __forceinline__ v8i load_fragB(const unsigned char* colPtr,
                                          int k0, int half) {
    const unsigned char* bp = colPtr + k0 + half * 16;
    v4i q0 = *reinterpret_cast<const v4i*>(bp);
    v4i q1 = *reinterpret_cast<const v4i*>(bp + 32);
    return (v8i){q0[0], q0[1], q0[2], q0[3], q1[0], q1[1], q1[2], q1[3]};
}

__global__ void k_gram_wmma(const unsigned char* __restrict__ Mq,
                            int* __restrict__ G) {
    const int lane = threadIdx.x;
    const int half = lane >> 4;
    const int m    = lane & 15;
    const int bm   = blockIdx.y * 2;
    const int bn   = (blockIdx.x * 4 + threadIdx.y) * 2;

    const unsigned char* Arow0 = Mq + (size_t)(bm * 16 + m) * KDIM;
    const unsigned char* Arow1 = Arow0 + (size_t)16 * KDIM;
    const unsigned char* Bcol0 = Mq + (size_t)(bn * 16 + m) * KDIM;
    const unsigned char* Bcol1 = Bcol0 + (size_t)16 * KDIM;

    v8i acc00 = {0,0,0,0,0,0,0,0};
    v8i acc01 = {0,0,0,0,0,0,0,0};
    v8i acc10 = {0,0,0,0,0,0,0,0};
    v8i acc11 = {0,0,0,0,0,0,0,0};

    for (int k0 = 0; k0 < KDIM; k0 += 64) {
        v8i a0 = load_fragA(Arow0, k0, half);
        v8i a1 = load_fragA(Arow1, k0, half);
        v8i b0 = load_fragB(Bcol0, k0, half);
        v8i b1 = load_fragB(Bcol1, k0, half);
        acc00 = __builtin_amdgcn_wmma_i32_16x16x64_iu8(false, a0, false, b0, acc00, false, false);
        acc01 = __builtin_amdgcn_wmma_i32_16x16x64_iu8(false, a0, false, b1, acc01, false, false);
        acc10 = __builtin_amdgcn_wmma_i32_16x16x64_iu8(false, a1, false, b0, acc10, false, false);
        acc11 = __builtin_amdgcn_wmma_i32_16x16x64_iu8(false, a1, false, b1, acc11, false, false);
    }

    const int r0 = bm * 16 + half * 8;
    const int c0 = bn * 16 + m;
    #pragma unroll
    for (int r = 0; r < 8; ++r) {
        G[(size_t)(r0 + r)      * NPAD + c0]      = acc00[r];
        G[(size_t)(r0 + r)      * NPAD + c0 + 16] = acc01[r];
        G[(size_t)(r0 + 16 + r) * NPAD + c0]      = acc10[r];
        G[(size_t)(r0 + 16 + r) * NPAD + c0 + 16] = acc11[r];
    }
}

__global__ void k_nms(const int* __restrict__ G,
                      const int* __restrict__ areaI,
                      const int* __restrict__ labels,
                      const float* __restrict__ scores,
                      const int* __restrict__ order,
                      float* __restrict__ sd) {
    const int tid = threadIdx.x;
    __shared__ int   ord[NPAD];
    __shared__ int   lab[NPAD];
    __shared__ float ar[NPAD];
    __shared__ float cmp[NPAD];

    int o = (tid < NINST) ? order[tid] : 0;
    ord[tid] = o;
    lab[tid] = (tid < NINST) ? labels[o] : -1;
    ar[tid]  = (tid < NINST) ? (float)areaI[o] : 0.0f;
    __syncthreads();

    float cm = 0.0f;
    if (tid < NINST) {
        const int oj = ord[tid];
        const float aj = ar[tid];
        const int lj = lab[tid];
        for (int i = 0; i < tid; ++i) {
            if (lab[i] == lj) {
                float inter = (float)G[(size_t)ord[i] * NPAD + oj];
                float uni   = ar[i] + aj - inter;
                cm = fmaxf(cm, inter / uni);
            }
        }
    }
    cmp[tid] = cm;
    __syncthreads();

    if (tid < NINST) {
        const int oj = ord[tid];
        const float aj = ar[tid];
        const int lj = lab[tid];
        float mn = FLT_MAX;
        for (int i = 0; i < NINST; ++i) {
            float d = 0.0f;
            if (i < tid && lab[i] == lj) {
                float inter = (float)G[(size_t)ord[i] * NPAD + oj];
                float uni   = ar[i] + aj - inter;
                d = inter / uni;
            }
            float ci = cmp[i];
            float t  = expf(-SIGMA_F * (d * d - ci * ci));
            mn = fminf(mn, t);
        }
        sd[tid] = scores[oj] * mn;
    }
}

__global__ void k_sort2_out(const float* __restrict__ sd,
                            const int* __restrict__ order,
                            const int* __restrict__ labels,
                            float* __restrict__ out,
                            int* __restrict__ keep) {
    const int tid = threadIdx.x;
    __shared__ float ss[NPAD];
    __shared__ int   o2[NPAD];
    ss[tid] = (tid < NINST) ? sd[tid] : -FLT_MAX;
    __syncthreads();
    if (tid < NINST) {
        const float sj = ss[tid];
        int rank = 0;
        for (int k = 0; k < NINST; ++k) {
            float sk = ss[k];
            rank += (sk > sj) || (sk == sj && k < tid);
        }
        o2[rank] = tid;
    }
    __syncthreads();
    if (tid < NINST) {
        int j  = o2[tid];
        int ki = order[j];
        out[tid]              = ss[j];
        out[NINST + tid]      = (float)labels[ki];
        out[2 * NINST + (size_t)NINST * HW + tid] = (float)ki;
        keep[tid] = ki;
    }
}

__global__ void k_mask_out(const unsigned char* __restrict__ Mq,
                           const int* __restrict__ keep,
                           float* __restrict__ out_masks) {
    const int r  = blockIdx.y;
    const int hw = blockIdx.x * 256 + threadIdx.x;
    if (hw < HW) {
        const int ki = keep[r];
        out_masks[(size_t)r * HW + hw] = (float)Mq[(size_t)ki * KDIM + hw];
    }
}

extern "C" void kernel_launch(void* const* d_in, const int* in_sizes, int n_in,
                              void* d_out, int out_size, void* d_ws, size_t ws_size,
                              hipStream_t stream) {
    const float* mask_preds = (const float*)d_in[0];
    const int*   labels     = (const int*)d_in[1];
    const float* scores     = (const float*)d_in[2];
    float* out = (float*)d_out;

    char* ws = (char*)d_ws;
    size_t off = 0;
    unsigned char* Mq = (unsigned char*)(ws + off); off += (size_t)NPAD * KDIM;
    int*   G      = (int*)(ws + off);   off += (size_t)NPAD * NPAD * sizeof(int);
    int*   areaI  = (int*)(ws + off);   off += NPAD * sizeof(int);
    int*   order  = (int*)(ws + off);   off += NPAD * sizeof(int);
    int*   keep   = (int*)(ws + off);   off += NPAD * sizeof(int);
    float* sd     = (float*)(ws + off); off += NPAD * sizeof(float);

    k_threshold<<<dim3(NPAD), dim3(256), 0, stream>>>(mask_preds, Mq, areaI);

    k_sort_scores<<<dim3(1), dim3(NPAD), 0, stream>>>(scores, order);

    k_gram_wmma<<<dim3(4, 16), dim3(32, 4), 0, stream>>>(Mq, G);

    k_nms<<<dim3(1), dim3(NPAD), 0, stream>>>(G, areaI, labels, scores, order, sd);

    k_sort2_out<<<dim3(1), dim3(NPAD), 0, stream>>>(sd, order, labels, out, keep);

    k_mask_out<<<dim3((HW + 255) / 256, NINST), dim3(256), 0, stream>>>(
        Mq, keep, out + 2 * NINST);
}
